// MultiHeadDiffAttn_3934190043396
// MI455X (gfx1250) — matrix operand order repaired and checked by reading, not compiled, not run on hardware
//
#include <hip/hip_runtime.h>
#include <math.h>

typedef __attribute__((ext_vector_type(16))) __bf16 v16bf;
typedef __attribute__((ext_vector_type(8)))  float  v8f;

#define LAMBDA_INIT 0.2f

// ---------- helpers ----------
__device__ __forceinline__ unsigned short f2bf(float f) {
    union { float f; unsigned u; } x; x.f = f;
    unsigned u = x.u;
    unsigned r = (u + 0x7FFFu + ((u >> 16) & 1u)) >> 16;   // RNE
    return (unsigned short)r;
}

union uv16 { v16bf v; uint4 q[2]; unsigned u[8]; };

__device__ __forceinline__ void async_ld_b128(const unsigned short* gsrc, void* ldst) {
    unsigned l = (unsigned)(uintptr_t)ldst;   // low 32 bits = LDS offset
    asm volatile("global_load_async_to_lds_b128 %0, %1, off"
                 :: "v"(l), "v"(gsrc) : "memory");
}
__device__ __forceinline__ void wait_async0() {
#if __has_builtin(__builtin_amdgcn_s_wait_asynccnt)
    __builtin_amdgcn_s_wait_asynccnt(0);
#else
    asm volatile("s_wait_asynccnt 0x0" ::: "memory");
#endif
}

// ---------- conversion kernels ----------
__global__ void convert_bf16_kernel(const float* __restrict__ src,
                                    unsigned short* __restrict__ dst, int n) {
    int i = blockIdx.x * blockDim.x + threadIdx.x;
    if (i < n) dst[i] = f2bf(src[i]);
}

// src [K][N] fp32 -> dst [N][K] bf16
__global__ void convert_transpose_bf16_kernel(const float* __restrict__ src,
                                              unsigned short* __restrict__ dst,
                                              int K, int N) {
    int i = blockIdx.x * blockDim.x + threadIdx.x;
    if (i < K * N) {
        int k = i / N, n = i - k * N;
        dst[(size_t)n * K + k] = f2bf(src[i]);
    }
}

// ---------- repack qkv (bf16 [B*T][3C]) into q, k (head-major) and V^T ----------
__global__ void repack_qkv_kernel(const unsigned short* __restrict__ qkv,
                                  unsigned short* __restrict__ qs,
                                  unsigned short* __restrict__ ks,
                                  unsigned short* __restrict__ vT) {
    constexpr int T = 1024, C = 1024, H = 16;
    size_t idx = (size_t)blockIdx.x * blockDim.x + threadIdx.x; // over B*T*C
    int c = (int)(idx & (C - 1));
    size_t bt = idx >> 10;          // token index
    int t = (int)(bt & (T - 1));
    int b = (int)(bt >> 10);
    const unsigned short* row = qkv + bt * (3 * C);
    int hh = c >> 5, d = c & 31;
    size_t qk = (((size_t)(b * 2 * H + hh)) * T + t) * 32 + d;
    qs[qk] = row[c];
    ks[qk] = row[C + c];
    int h = c >> 6, vd = c & 63;
    vT[(((size_t)(b * H + h)) * 64 + vd) * T + t] = row[2 * C + c];
}

// ---------- bf16 WMMA GEMM: A[M][K] * B^T[N][K] -> C[M][N] ----------
// Block tile 64x128, K-step 32, 8 waves (2x4), each wave 32x32.
// Double-buffered LDS filled with async-to-LDS loads (ASYNCcnt).
template <bool OUT_BF16>
__global__ __launch_bounds__(256)
void gemm_bf16_wmma_kernel(const unsigned short* __restrict__ A,
                           const unsigned short* __restrict__ BT,
                           void* __restrict__ Cout, int M, int N, int K) {
    __shared__ alignas(16) unsigned short lds_a[2][64][32];
    __shared__ alignas(16) unsigned short lds_b[2][128][32];
    const int m0 = blockIdx.y * 64;
    const int n0 = blockIdx.x * 128;
    const int tid = threadIdx.x;
    const int wave = tid >> 5, lane = tid & 31;
    const int wr = wave & 1, wc = wave >> 1;
    const int ln = lane & 15, lh = lane >> 4;

    // per-thread staging slots
    const int arow = tid >> 2, ac8 = (tid & 3) * 8;    // A: 64 rows x 32, 16B/thread
    const int brow = tid >> 1, bc16 = (tid & 1) * 16;  // B: 128 rows x 32, 32B/thread
    const unsigned short* agp = A + (size_t)(m0 + arow) * K + ac8;
    const unsigned short* bgp = BT + (size_t)(n0 + brow) * K + bc16;

    v8f acc[2][2];
    for (int i = 0; i < 2; ++i)
        for (int j = 0; j < 2; ++j)
            for (int r = 0; r < 8; ++r) acc[i][j][r] = 0.0f;

    auto stage = [&](int buf, int k0) {
        async_ld_b128(agp + k0,     &lds_a[buf][arow][ac8]);
        async_ld_b128(bgp + k0,     &lds_b[buf][brow][bc16]);
        async_ld_b128(bgp + k0 + 8, &lds_b[buf][brow][bc16 + 8]);
    };

    stage(0, 0);
    const int ntiles = K / 32;
    for (int it = 0; it < ntiles; ++it) {
        const int cur = it & 1;
        const int k0 = it * 32;
        wait_async0();          // my async loads for tile `it` landed in LDS
        __syncthreads();        // everyone's landed; prev compute reads done
        if (it + 1 < ntiles) stage(cur ^ 1, k0 + 32);

        uv16 af[2], bf[2];
        for (int i = 0; i < 2; ++i) { // A frag: lane m = ln, elems: d=lh*8+e , 16+lh*8+e
            int m = wr * 32 + i * 16 + ln;
            af[i].q[0] = *(const uint4*)&lds_a[cur][m][lh * 8];
            af[i].q[1] = *(const uint4*)&lds_a[cur][m][16 + lh * 8];
        }
        for (int j = 0; j < 2; ++j) { // B frag: lane n = ln, elems: k=lh*8+e , 16+lh*8+e
            int n = wc * 32 + j * 16 + ln;
            bf[j].q[0] = *(const uint4*)&lds_b[cur][n][lh * 8];
            bf[j].q[1] = *(const uint4*)&lds_b[cur][n][16 + lh * 8];
        }
        for (int i = 0; i < 2; ++i)
            for (int j = 0; j < 2; ++j)
                acc[i][j] = __builtin_amdgcn_wmma_f32_16x16x32_bf16(
                    false, af[i].v, false, bf[j].v, (short)0, acc[i][j], false, false);
    }

    for (int i = 0; i < 2; ++i)
        for (int j = 0; j < 2; ++j) {
            int n = n0 + wc * 32 + j * 16 + ln;
            for (int r = 0; r < 8; ++r) {
                int m = m0 + wr * 32 + i * 16 + lh * 8 + r;
                float v = acc[i][j][r];
                if (OUT_BF16)
                    ((unsigned short*)Cout)[(size_t)m * N + n] = f2bf(v);
                else
                    ((float*)Cout)[(size_t)m * N + n] = v;
            }
        }
}

// ---------- differential flash attention (transposed-score formulation) ----------
// grid: (T/64, H, B); 8 waves: wave = (qt<<1)|sub ; each wave: 16 q rows, one sub-head.
__global__ __launch_bounds__(256)
void diff_attn_kernel(const unsigned short* __restrict__ qs,
                      const unsigned short* __restrict__ ks,
                      const unsigned short* __restrict__ vT,
                      const float* __restrict__ lq1, const float* __restrict__ lq2,
                      const float* __restrict__ lk1, const float* __restrict__ lk2,
                      const float* __restrict__ gamma,
                      unsigned short* __restrict__ attn_out /* bf16 [B*T][C] */) {
    constexpr int T = 1024, H = 16, HD = 32, VD = 64;
    __shared__ float lds_o[2][64][VD + 1];
    __shared__ float lds_lambda;
    __shared__ float lds_gamma[VD];

    const int qb0 = blockIdx.x * 64;
    const int h = blockIdx.y;
    const int b = blockIdx.z;
    const int tid = threadIdx.x;
    const int lane = tid & 31, wave = tid >> 5;
    const int sub = wave & 1, qt = wave >> 1;
    const int ln = lane & 15, lh = lane >> 4;

    if (tid < 32) { // lambda_full = exp(lq1.lk1) - exp(lq2.lk2) + lambda_init
        float p1 = lq1[lane] * lk1[lane];
        float p2 = lq2[lane] * lk2[lane];
        for (int m = 16; m >= 1; m >>= 1) {
            p1 += __shfl_xor(p1, m, 32);
            p2 += __shfl_xor(p2, m, 32);
        }
        if (lane == 0) lds_lambda = __expf(p1) - __expf(p2) + LAMBDA_INIT;
    }
    if (tid < VD) lds_gamma[tid] = gamma[tid];

    const int hh = 2 * h + sub;
    const int qrow0 = qb0 + qt * 16;
    const unsigned short* qbase = qs + (((size_t)(b * 2 * H + hh)) * T + qrow0) * HD;
    const unsigned short* kbase = ks + ((size_t)(b * 2 * H + hh)) * T * HD;
    const unsigned short* vbase = vT + ((size_t)(b * H + h)) * VD * T;

    // Q^T B-fragment: lane col n=ln -> token qrow0+ln; elems e -> d = lh*8+e , 16+lh*8+e
    uv16 qf;
    {
        const uint4* p = (const uint4*)(qbase + (size_t)ln * HD + lh * 8);
        qf.q[0] = p[0]; qf.q[1] = p[2];
    }

    float mprev = -INFINITY, lsum = 0.0f;
    v8f o[4];
    for (int c = 0; c < 4; ++c)
        for (int r = 0; r < 8; ++r) o[c][r] = 0.0f;

    const float scale = 1.0f / 32.0f;       // (hd^-0.5)*(1/sqrt(hd)) = 1/hd
    const int qtok = qrow0 + ln;            // this lane's query column
    const int smax = qrow0 + 16;

    for (int s0 = 0; s0 < smax; s0 += 32) {
        // prefetch next chunk's K rows and V rows into cache
        if (s0 + 32 < smax) {
            __builtin_prefetch(kbase + (size_t)(s0 + 32 + ln) * HD, 0, 3);
            __builtin_prefetch(vbase + (size_t)ln * T + s0 + 32, 0, 3);
        }
        // two S^T tiles: A = K rows (16 x 32), B = Q^T
        float pva[8], pvb[8];
        float cmax = -INFINITY;
        for (int j = 0; j < 2; ++j) {
            int srow = s0 + j * 16 + ln;
            uv16 kf;
            const unsigned short* kr = kbase + (size_t)srow * HD;
            kf.q[0] = *(const uint4*)(kr + lh * 8);
            kf.q[1] = *(const uint4*)(kr + 16 + lh * 8);
            v8f z;
            for (int r = 0; r < 8; ++r) z[r] = 0.0f;
            v8f st = __builtin_amdgcn_wmma_f32_16x16x32_bf16(
                false, kf.v, false, qf.v, (short)0, z, false, false);
            for (int r = 0; r < 8; ++r) {
                int stok = s0 + j * 16 + lh * 8 + r;
                float v = (stok <= qtok) ? st[r] * scale : -INFINITY;
                if (j == 0) pva[r] = v; else pvb[r] = v;
                cmax = fmaxf(cmax, v);
            }
        }
        cmax = fmaxf(cmax, __shfl_xor(cmax, 16, 32));
        float mnew = fmaxf(mprev, cmax);
        float alpha = __expf(mprev - mnew);
        float rowsum = 0.0f;
        for (int r = 0; r < 8; ++r) {
            pva[r] = __expf(pva[r] - mnew);
            pvb[r] = __expf(pvb[r] - mnew);
            rowsum += pva[r] + pvb[r];
        }
        rowsum += __shfl_xor(rowsum, 16, 32);
        lsum = lsum * alpha + rowsum;
        mprev = mnew;
        for (int c = 0; c < 4; ++c)
            for (int r = 0; r < 8; ++r) o[c][r] *= alpha;

        // build P^T B-fragment (32x16): keep tile==lh, exchange tile==1-lh across halves
        unsigned keep[4], send[4];
        for (int i = 0; i < 4; ++i) {
            float k0f = lh ? pvb[2 * i]     : pva[2 * i];
            float k1f = lh ? pvb[2 * i + 1] : pva[2 * i + 1];
            float s0f = lh ? pva[2 * i]     : pvb[2 * i];
            float s1f = lh ? pva[2 * i + 1] : pvb[2 * i + 1];
            keep[i] = (unsigned)f2bf(k0f) | ((unsigned)f2bf(k1f) << 16);
            send[i] = (unsigned)f2bf(s0f) | ((unsigned)f2bf(s1f) << 16);
        }
        unsigned ex[4];
        for (int i = 0; i < 4; ++i) ex[i] = __shfl_xor(send[i], 16, 32);
        uv16 pf;
        for (int i = 0; i < 4; ++i) {
            pf.u[i]     = lh ? ex[i]   : keep[i];
            pf.u[4 + i] = lh ? keep[i] : ex[i];
        }

        // O^T += V^T * P^T : 4 vd-chunks of 16 rows, K = 32 s-values
        for (int c = 0; c < 4; ++c) {
            const unsigned short* vr = vbase + (size_t)(c * 16 + ln) * T + s0;
            uv16 vf;
            vf.q[0] = *(const uint4*)(vr + lh * 16);
            vf.q[1] = *(const uint4*)(vr + lh * 16 + 8);
            o[c] = __builtin_amdgcn_wmma_f32_16x16x32_bf16(
                false, vf.v, false, pf.v, (short)0, o[c], false, false);
        }
    }

    float inv_l = 1.0f / lsum;
    for (int c = 0; c < 4; ++c)
        for (int r = 0; r < 8; ++r)
            lds_o[sub][qt * 16 + ln][c * 16 + lh * 8 + r] = o[c][r] * inv_l;
    __syncthreads();

    // combine sub-heads, RMSNorm over 64, scale, write bf16
    {
        float lam = lds_lambda;
        int ql = tid >> 2;
        int vd0 = (tid & 3) * 16;
        float comb[16], ss = 0.0f;
        for (int i = 0; i < 16; ++i) {
            float v = lds_o[0][ql][vd0 + i] - lam * lds_o[1][ql][vd0 + i];
            comb[i] = v;
            ss += v * v;
        }
        ss += __shfl_xor(ss, 1, 32);
        ss += __shfl_xor(ss, 2, 32);
        float rms = rsqrtf(ss * (1.0f / 64.0f) + 1e-5f);
        const float c1 = 1.0f - LAMBDA_INIT;
        size_t row = (size_t)b * T + qb0 + ql;
        unsigned short* dst = attn_out + row * 1024 + h * 64 + vd0;
        for (int i = 0; i < 16; ++i)
            dst[i] = f2bf(comb[i] * rms * lds_gamma[vd0 + i] * c1);
    }
}

// ---------- host launch ----------
extern "C" void kernel_launch(void* const* d_in, const int* in_sizes, int n_in,
                              void* d_out, int out_size, void* d_ws, size_t ws_size,
                              hipStream_t stream) {
    constexpr int B = 4, T = 1024, C = 1024, H = 16;
    const float* x      = (const float*)d_in[0];
    const float* w_attn = (const float*)d_in[1];
    const float* w_proj = (const float*)d_in[2];
    const float* lq1    = (const float*)d_in[3];
    const float* lq2    = (const float*)d_in[4];
    const float* lk1    = (const float*)d_in[5];
    const float* lk2    = (const float*)d_in[6];
    const float* gamma  = (const float*)d_in[7];
    float* out = (float*)d_out;

    char* ws = (char*)d_ws;
    size_t o = 0;
    unsigned short* x_bf = (unsigned short*)(ws + o); o += (size_t)B * T * C * 2;          // 8 MB
    unsigned short* wTa  = (unsigned short*)(ws + o); o += (size_t)3 * C * C * 2;          // 6 MB
    unsigned short* wTp  = (unsigned short*)(ws + o); o += (size_t)C * C * 2;              // 2 MB
    unsigned short* qkv  = (unsigned short*)(ws + o); o += (size_t)B * T * 3 * C * 2;      // 24 MB
    unsigned short* qsb  = (unsigned short*)(ws + o); o += (size_t)B * T * C * 2;          // 8 MB
    unsigned short* ksb  = (unsigned short*)(ws + o); o += (size_t)B * T * C * 2;          // 8 MB
    unsigned short* vTb  = (unsigned short*)(ws + o); o += (size_t)B * T * C * 2;          // 8 MB
    unsigned short* attn = (unsigned short*)(ws + o); o += (size_t)B * T * C * 2;          // 8 MB
    (void)ws_size; (void)in_sizes; (void)n_in; (void)out_size;

    // 1) convert / transpose to bf16
    convert_bf16_kernel<<<(B * T * C) / 256, 256, 0, stream>>>(x, x_bf, B * T * C);
    convert_transpose_bf16_kernel<<<(C * 3 * C) / 256, 256, 0, stream>>>(w_attn, wTa, C, 3 * C);
    convert_transpose_bf16_kernel<<<(C * C) / 256, 256, 0, stream>>>(w_proj, wTp, C, C);

    // 2) qkv = x @ w_attn  (bf16 WMMA, bf16 out, async double-buffered LDS)
    gemm_bf16_wmma_kernel<true><<<dim3(3 * C / 128, B * T / 64), 256, 0, stream>>>(
        x_bf, wTa, qkv, B * T, 3 * C, C);

    // 3) repack into head-major q/k and V^T
    repack_qkv_kernel<<<(B * T * C) / 256, 256, 0, stream>>>(qkv, qsb, ksb, vTb);

    // 4) differential flash attention + RMSNorm -> bf16 [B*T][C]
    diff_attn_kernel<<<dim3(T / 64, H, B), 256, 0, stream>>>(
        qsb, ksb, vTb, lq1, lq2, lk1, lk2, gamma, attn);

    // 5) out = attn @ w_proj (fp32 out)
    gemm_bf16_wmma_kernel<false><<<dim3(C / 128, B * T / 64), 256, 0, stream>>>(
        attn, wTp, out, B * T, C, C);
}
